// DMCG_NN_Pos_Fixed_56367150792727
// MI455X (gfx1250) — matrix operand order repaired and checked by reading, not compiled, not run on hardware
//
// DMCG GNN forward for MI455X (gfx1250) — BF16 WMMA pipeline, v3b.
//  - B tile staged to LDS via GLOBAL_LOAD_ASYNC_TO_LDS_B128 when available
//    (ASYNCcnt path; falls back to load+ds_store), double-buffered.
//  - A fragments ping-pong in a 2-entry register array (no v_mov chain).
//  - All 4 B fragments loaded into distinct regs -> one s_wait_dscnt per K-step.
//  - Residual adds fused into layer-3 GEMM epilogue.
//
// Assumed d_in leaf order (jax pytree, dict keys sorted, {"b","w"} per layer):
//   [0] conforms
//   [1 + s*18 + {0:edge,6:glob,12:node} + l*2 + {0:b,1:w}]  for gnn step s, layer l
//   [217..222] posee (b0,w0,b1,w1,b2,w2), [223..228] poseg, [229..234] posen
#include <hip/hip_runtime.h>

#define B_G   32
#define N_A   64
#define LAT   256
#define HID   512
#define E_G   4032                 // edges per graph (64*63)
#define NE    (B_G*E_G)            // 129024
#define NNODE (B_G*N_A)            // 2048
#define MT_E  (NE/16)              // 8064 row tiles (edges)
#define MT_N  (NNODE/16)           // 128  row tiles (nodes)
#define MT_G  (B_G/16)             // 2    row tiles (graphs)

typedef unsigned short u16;
typedef __attribute__((ext_vector_type(16))) __bf16 v16bf;
typedef __attribute__((ext_vector_type(8)))  float  v8f;
typedef int v4i __attribute__((vector_size(16)));   // matches builtin param pointee

#if defined(__gfx1250__) && __has_builtin(__builtin_amdgcn_global_load_async_to_lds_b128) && \
    __has_builtin(__builtin_amdgcn_s_wait_asynccnt)
#define HAVE_ASYNC 1
#define AS1 __attribute__((address_space(1)))
#define AS3 __attribute__((address_space(3)))
#else
#define HAVE_ASYNC 0
#endif

union Frag16 { u16 u[16]; v16bf v; };

__device__ __forceinline__ u16 f2bf(float f) {   // f32 -> bf16, round-nearest-even
  unsigned u = __float_as_uint(f);
  u += 0x7FFFu + ((u >> 16) & 1u);
  return (u16)(u >> 16);
}

// ---------------------------------------------------------------------------
// Distances: d[e] = |pos[src]-pos[dst]|, i-major edge order.
// ---------------------------------------------------------------------------
__global__ void k_dist(const float* __restrict__ pos, float* __restrict__ d) {
  int e = blockIdx.x * blockDim.x + threadIdx.x;
  if (e >= NE) return;
  int g = e / E_G, r = e % E_G, i = r / 63, jj = r % 63, j = jj + (jj >= i);
  long a = (long)(g * N_A + i) * 3, b = (long)(g * N_A + j) * 3;
  float dx = pos[a] - pos[b], dy = pos[a + 1] - pos[b + 1], dz = pos[a + 2] - pos[b + 2];
  d[e] = sqrtf(dx * dx + dy * dy + dz * dz);
}

// ---------------------------------------------------------------------------
// Packing kernels: one wave builds one 512-elem bf16 fragment (32 lanes x 32B).
// A-fragment (16x32, MxK): lane row = lane&15; K halves = (lane>>4)*8 + {0..7, +16}.
// B-fragment (32x16, KxN): lane col = lane&15; K halves = (lane>>4)*8 + {0..7, +16}.
// ---------------------------------------------------------------------------
__global__ void k_pack_w(const float* __restrict__ W, u16* __restrict__ dst,
                         int KT, int Kreal, int Ncols, long frags) {
  int wave = threadIdx.x >> 5, lane = threadIdx.x & 31;
  long frag = (long)blockIdx.x * 8 + wave;
  if (frag >= frags) return;
  int nt = (int)(frag / KT), kt = (int)(frag % KT);
  int n = nt * 16 + (lane & 15);
  int kbase = kt * 32 + ((lane >> 4) << 3);
  Frag16 f;
#pragma unroll
  for (int h = 0; h < 16; ++h) {
    int k = kbase + (h < 8 ? h : h + 8);
    f.u[h] = f2bf((k < Kreal) ? W[(long)k * Ncols + n] : 0.0f);
  }
  *(v16bf*)(dst + (frag << 9) + (long)lane * 16) = f.v;
}

__global__ void k_pack_mat(const float* __restrict__ src, u16* __restrict__ dst,
                           long frags, int KT, int Kreal) {
  int wave = threadIdx.x >> 5, lane = threadIdx.x & 31;
  long frag = (long)blockIdx.x * 8 + wave;
  if (frag >= frags) return;
  int mt = (int)(frag / KT), kt = (int)(frag % KT);
  int row = mt * 16 + (lane & 15);
  int koff = (lane >> 4) << 3;
  Frag16 f;
#pragma unroll
  for (int h = 0; h < 16; ++h) {
    int k = kt * 32 + koff + (h < 8 ? h : h + 8);
    f.u[h] = f2bf((k < Kreal) ? src[(long)row * Kreal + k] : 0.0f);
  }
  *(v16bf*)(dst + (frag << 9) + (long)lane * 16) = f.v;
}

// concat([edge, node[ROWS], node[COLS], glob[graph]]) -> A fragments, K=1024
__global__ void k_pack_edge(const float* __restrict__ eo, const float* __restrict__ no,
                            const float* __restrict__ go, u16* __restrict__ dst) {
  int wave = threadIdx.x >> 5, lane = threadIdx.x & 31;
  long frag = (long)blockIdx.x * 8 + wave;
  if (frag >= (long)MT_E * 32) return;
  int mt = (int)(frag >> 5), kt = (int)(frag & 31);
  int e = mt * 16 + (lane & 15);
  int g = e / E_G, r = e % E_G, i = r / 63, jj = r % 63, j = jj + (jj >= i);
  long rE = (long)e * LAT, rS = (long)(g * N_A + i) * LAT,
       rT = (long)(g * N_A + j) * LAT, rG = (long)g * LAT;
  int koff = (lane >> 4) << 3;
  Frag16 f;
#pragma unroll
  for (int h = 0; h < 16; ++h) {
    int k = kt * 32 + koff + (h < 8 ? h : h + 8);
    float v;
    if (k < 256)      v = eo[rE + k];
    else if (k < 512) v = no[rS + k - 256];
    else if (k < 768) v = no[rT + k - 512];
    else              v = go[rG + k - 768];
    f.u[h] = f2bf(v);
  }
  *(v16bf*)(dst + (frag << 9) + (long)lane * 16) = f.v;
}

// concat([node, sent, recv, glob[graph]]) -> A fragments, K=1024
__global__ void k_pack_node(const float* __restrict__ no, const float* __restrict__ se,
                            const float* __restrict__ re, const float* __restrict__ go,
                            u16* __restrict__ dst) {
  int wave = threadIdx.x >> 5, lane = threadIdx.x & 31;
  long frag = (long)blockIdx.x * 8 + wave;
  if (frag >= (long)MT_N * 32) return;
  int mt = (int)(frag >> 5), kt = (int)(frag & 31);
  int nd = mt * 16 + (lane & 15);
  long rN = (long)nd * LAT, rG = (long)(nd >> 6) * LAT;
  int koff = (lane >> 4) << 3;
  Frag16 f;
#pragma unroll
  for (int h = 0; h < 16; ++h) {
    int k = kt * 32 + koff + (h < 8 ? h : h + 8);
    float v;
    if (k < 256)      v = no[rN + k];
    else if (k < 512) v = se[rN + k - 256];
    else if (k < 768) v = re[rN + k - 512];
    else              v = go[rG + k - 768];
    f.u[h] = f2bf(v);
  }
  *(v16bf*)(dst + (frag << 9) + (long)lane * 16) = f.v;
}

// concat([glob, gn, ge]) -> A fragments, K=768
__global__ void k_pack_glob(const float* __restrict__ go, const float* __restrict__ gn,
                            const float* __restrict__ ge, u16* __restrict__ dst) {
  int wave = threadIdx.x >> 5, lane = threadIdx.x & 31;
  long frag = (long)blockIdx.x * 8 + wave;
  if (frag >= (long)MT_G * 24) return;
  int mt = (int)(frag / 24), kt = (int)(frag % 24);
  int g = mt * 16 + (lane & 15);
  long rG = (long)g * LAT;
  int koff = (lane >> 4) << 3;
  Frag16 f;
#pragma unroll
  for (int h = 0; h < 16; ++h) {
    int k = kt * 32 + koff + (h < 8 ? h : h + 8);
    float v;
    if (k < 256)      v = go[rG + k];
    else if (k < 512) v = gn[rG + k - 256];
    else              v = ge[rG + k - 512];
    f.u[h] = f2bf(v);
  }
  *(v16bf*)(dst + (frag << 9) + (long)lane * 16) = f.v;
}

// ---------------------------------------------------------------------------
// WMMA GEMM: C[M,N] = act(A @ W + b). 8 waves/block, 16x64 per wave, 128x64/block.
// B tile (4KB/K-step) staged in LDS (async-to-LDS when available), double-buffered.
// WPK: epilogue re-packs output to next layer's bf16 A-fragments via LDS.
// RES: epilogue also does resid[i] += out (fused residual).
// ---------------------------------------------------------------------------
template <int RELU, int WF32, int WPK, int RES>
__launch_bounds__(256)
__global__ void k_gemm(const u16* __restrict__ A, const u16* __restrict__ W,
                       const float* __restrict__ bias,
                       float* __restrict__ outF, float* __restrict__ resid,
                       u16* __restrict__ outP,
                       int Mtiles, int KT, int Ncols, int KTnext) {
  __shared__ float ldsC[8 * 16 * 64];                 // epilogue re-pack staging
  __shared__ __align__(16) u16 bbuf[2][4 * 512];      // double-buffered B tile
  int tid = threadIdx.x;
  int wave = tid >> 5, lane = tid & 31;
  int mt = blockIdx.x * 8 + wave;
  const bool live = (mt < Mtiles);                    // wave-uniform
  if (!live) mt = Mtiles - 1;                         // tail waves duplicate, no store
  int cb = blockIdx.y * 64;

  // B staging plan: thread t moves 16B of fragment snt = t>>6 each K-step.
  int snt = tid >> 6;
  int sof = (tid & 63) * 8;
  const u16* wsrc = W + ((long)((cb >> 4) + snt) * KT) * 512 + sof;
  const u16* ap = A + ((long)mt * KT) * 512 + (long)lane * 16;

  v8f acc[4] = {};
  v16bf areg[2];

  // prologue: stage K-tile 0, load A-frag 0
#if HAVE_ASYNC
  __builtin_amdgcn_global_load_async_to_lds_b128(
      (AS1 v4i*)wsrc, (AS3 v4i*)&bbuf[0][snt * 512 + sof], 0, 0);
  areg[0] = *(const v16bf*)ap;
  __builtin_amdgcn_s_wait_asynccnt(0);
#else
  {
    uint4 stg0 = *(const uint4*)wsrc;
    areg[0] = *(const v16bf*)ap;
    *(uint4*)&bbuf[0][snt * 512 + sof] = stg0;
  }
#endif
  __syncthreads();

#pragma unroll 2
  for (int kt = 0; kt < KT; ++kt) {
    const bool more = (kt + 1 < KT);
#if HAVE_ASYNC
    if (more) {                                       // async fill of other buffer
      __builtin_amdgcn_global_load_async_to_lds_b128(
          (AS1 v4i*)(wsrc + (kt + 1) * 512),
          (AS3 v4i*)&bbuf[(kt + 1) & 1][snt * 512 + sof], 0, 0);
      areg[(kt + 1) & 1] = *(const v16bf*)(ap + 512);
      __builtin_prefetch(wsrc + (kt + 2) * 512, 0, 1);
    }
#else
    uint4 stgN;
    if (more) {
      stgN = *(const uint4*)(wsrc + (kt + 1) * 512);
      areg[(kt + 1) & 1] = *(const v16bf*)(ap + 512);
      __builtin_prefetch(wsrc + (kt + 2) * 512, 0, 1);
    }
#endif
    const u16* bb = &bbuf[kt & 1][lane * 16];
    v16bf b0 = *(const v16bf*)(bb);                   // batch all 8 ds_load_b128,
    v16bf b1 = *(const v16bf*)(bb + 512);             // one s_wait_dscnt total
    v16bf b2 = *(const v16bf*)(bb + 1024);
    v16bf b3 = *(const v16bf*)(bb + 1536);
    v16bf a_cur = areg[kt & 1];
    acc[0] = __builtin_amdgcn_wmma_f32_16x16x32_bf16(false, a_cur, false, b0, (short)0, acc[0], false, false);
    acc[1] = __builtin_amdgcn_wmma_f32_16x16x32_bf16(false, a_cur, false, b1, (short)0, acc[1], false, false);
    acc[2] = __builtin_amdgcn_wmma_f32_16x16x32_bf16(false, a_cur, false, b2, (short)0, acc[2], false, false);
    acc[3] = __builtin_amdgcn_wmma_f32_16x16x32_bf16(false, a_cur, false, b3, (short)0, acc[3], false, false);
#if !HAVE_ASYNC
    if (more) *(uint4*)&bbuf[(kt + 1) & 1][snt * 512 + sof] = stgN;
#endif
    if (more) ap += 512;
#if HAVE_ASYNC
    __builtin_amdgcn_s_wait_asynccnt(0);              // async fills visible pre-barrier
#endif
    __syncthreads();
  }

  int ch = lane & 15;                                 // C: column-in-tile
  int rhi = (lane >> 4) << 3;                         // C: row offset (0 or 8)
  float* myl = ldsC + wave * 1024;
#pragma unroll
  for (int nt = 0; nt < 4; ++nt) {
    int col = cb + nt * 16 + ch;
    float bv = bias[col];
#pragma unroll
    for (int r = 0; r < 8; ++r) {
      float v = acc[nt][r] + bv;
      if (RELU) v = fmaxf(v, 0.0f);
      if (WF32 && live) {
        long idx = (long)(mt * 16 + rhi + r) * Ncols + col;
        outF[idx] = v;
        if (RES) resid[idx] += v;                     // fused residual
      }
      if (WPK) myl[(rhi + r) * 64 + nt * 16 + ch] = v;
    }
  }
  if (WPK) {
    __builtin_amdgcn_fence(__ATOMIC_ACQ_REL, "wavefront");   // order in-wave LDS RAW
    int rowA = ch, koff = rhi;
#pragma unroll
    for (int kt2 = 0; kt2 < 2; ++kt2) {
      Frag16 f;
#pragma unroll
      for (int h = 0; h < 16; ++h) {
        int k = kt2 * 32 + koff + (h < 8 ? h : h + 8);
        f.u[h] = f2bf(myl[rowA * 64 + k]);
      }
      if (live) {
        long ktg = (long)(cb >> 5) + kt2;
        *(v16bf*)(outP + ((long)mt * KTnext + ktg) * 512 + (long)lane * 16) = f.v;
      }
    }
  }
}

// ---------------------------------------------------------------------------
// Structured segment sums + elementwise ops.
// ---------------------------------------------------------------------------
__global__ void k_sent(const float* __restrict__ e1, float* __restrict__ out) {
  int idx = blockIdx.x * blockDim.x + threadIdx.x;
  if (idx >= NNODE * LAT) return;
  int nd = idx >> 8, c = idx & 255, g = nd >> 6, i = nd & 63;
  long base = ((long)g * E_G + (long)i * 63) * LAT + c;
  float s = 0.f;
  for (int k = 0; k < 63; ++k) s += e1[base + (long)k * LAT];
  out[idx] = s;
}

__global__ void k_recv(const float* __restrict__ e1, float* __restrict__ out) {
  int idx = blockIdx.x * blockDim.x + threadIdx.x;
  if (idx >= NNODE * LAT) return;
  int nd = idx >> 8, c = idx & 255, g = nd >> 6, j = nd & 63;
  float s = 0.f;
  for (int i = 0; i < N_A; ++i) {
    if (i == j) continue;
    int p = (j < i) ? j : j - 1;
    s += e1[((long)g * E_G + (long)i * 63 + p) * LAT + c];
  }
  out[idx] = s;
}

__global__ void k_gsum(const float* __restrict__ src, float* __restrict__ out) {
  int idx = blockIdx.x * blockDim.x + threadIdx.x;          // [2048,256] -> [32,256]
  if (idx >= B_G * LAT) return;
  int g = idx >> 8, c = idx & 255;
  float s = 0.f;
  for (int i = 0; i < N_A; ++i) s += src[((long)g * N_A + i) * LAT + c];
  out[idx] = s;
}

__global__ void k_copy(float* __restrict__ d, const float* __restrict__ s, long n) {
  long i = (long)blockIdx.x * blockDim.x + threadIdx.x;
  if (i < n) d[i] = s[i];
}

// ---------------------------------------------------------------------------
// Host orchestration
// ---------------------------------------------------------------------------
struct MLP { const float *w0, *b0, *w1, *b1, *w2, *b2; u16 *p0, *p1, *p2; int K0; };

extern "C" void kernel_launch(void* const* d_in, const int* in_sizes, int n_in,
                              void* d_out, int out_size, void* d_ws, size_t ws_size,
                              hipStream_t stream) {
  (void)in_sizes; (void)n_in; (void)out_size; (void)ws_size;
  const float* conforms = (const float*)d_in[0];

  char* base = (char*)d_ws;
  size_t off = 0;
  auto alloc = [&](size_t bytes) -> void* {
    void* p = (void*)(base + off);
    off += (bytes + 255) & ~(size_t)255;
    return p;
  };
  float* dD   = (float*)alloc((size_t)NE * 4);
  float* node = (float*)alloc((size_t)NNODE * LAT * 4);
  float* edge = (float*)alloc((size_t)NE * LAT * 4);
  float* glob = (float*)alloc((size_t)B_G * LAT * 4);
  float* e1   = (float*)alloc((size_t)NE * LAT * 4);
  float* n1   = (float*)alloc((size_t)NNODE * LAT * 4);
  float* g1v  = (float*)alloc((size_t)B_G * LAT * 4);
  float* sent = (float*)alloc((size_t)NNODE * LAT * 4);
  float* recv = (float*)alloc((size_t)NNODE * LAT * 4);
  float* gn   = (float*)alloc((size_t)B_G * LAT * 4);
  float* ge   = (float*)alloc((size_t)B_G * LAT * 4);
  u16* Apk = (u16*)alloc((size_t)MT_E * 32 * 512 * 2);   // biggest A (edge concat)
  u16* H1  = (u16*)alloc((size_t)MT_E * 16 * 512 * 2);   // hidden ping
  u16* H2  = Apk;                                        // hidden pong aliases Apk (Apk dead after L1)
  u16* Wp  = (u16*)alloc((size_t)34750464 * 2);          // all packed weights

  size_t woff = 0;
  auto packW = [&](const float* W, int Kreal, int Ncols) -> u16* {
    int KT = (Kreal + 31) / 32, NT = Ncols / 16;
    u16* dst = Wp + woff;
    woff += (size_t)NT * KT * 512;
    long frags = (long)NT * KT;
    k_pack_w<<<dim3((unsigned)((frags + 7) / 8)), 256, 0, stream>>>(W, dst, KT, Kreal, Ncols, frags);
    return dst;
  };
  auto buildMLP = [&](int bidx, int K0) -> MLP {
    MLP m;
    m.b0 = (const float*)d_in[bidx + 0]; m.w0 = (const float*)d_in[bidx + 1];
    m.b1 = (const float*)d_in[bidx + 2]; m.w1 = (const float*)d_in[bidx + 3];
    m.b2 = (const float*)d_in[bidx + 4]; m.w2 = (const float*)d_in[bidx + 5];
    m.K0 = K0;
    m.p0 = packW(m.w0, K0, HID);
    m.p1 = packW(m.w1, HID, HID);
    m.p2 = packW(m.w2, HID, LAT);
    return m;
  };

  MLP mE[12], mN[12], mG[12];
  for (int st = 0; st < 12; ++st) {
    int b0 = 1 + st * 18;
    mE[st] = buildMLP(b0 + 0, 4 * LAT);
    mG[st] = buildMLP(b0 + 6, 3 * LAT);
    mN[st] = buildMLP(b0 + 12, 4 * LAT);
  }
  MLP mPosee = buildMLP(217, 1);
  MLP mPoseg = buildMLP(223, E_G);
  MLP mPosen = buildMLP(229, N_A - 1);

  auto runMLP = [&](const MLP& m, const u16* Ain, int Mtiles, float* outF, float* res) {
    int KT0 = (m.K0 + 31) / 32;
    dim3 gh((unsigned)((Mtiles + 7) / 8), HID / 64);
    k_gemm<1, 0, 1, 0><<<gh, 256, 0, stream>>>(Ain, m.p0, m.b0, (float*)nullptr,
                                               (float*)nullptr, H1, Mtiles, KT0, HID, HID / 32);
    k_gemm<1, 0, 1, 0><<<gh, 256, 0, stream>>>(H1, m.p1, m.b1, (float*)nullptr,
                                               (float*)nullptr, H2, Mtiles, HID / 32, HID, HID / 32);
    dim3 go((unsigned)((Mtiles + 7) / 8), LAT / 64);
    if (res)
      k_gemm<0, 1, 0, 1><<<go, 256, 0, stream>>>(H2, m.p2, m.b2, outF, res,
                                                 (u16*)nullptr, Mtiles, HID / 32, LAT, 0);
    else
      k_gemm<0, 1, 0, 0><<<go, 256, 0, stream>>>(H2, m.p2, m.b2, outF, (float*)nullptr,
                                                 (u16*)nullptr, Mtiles, HID / 32, LAT, 0);
  };

  // distances
  k_dist<<<(NE + 255) / 256, 256, 0, stream>>>(conforms, dD);

  // position encoders
  k_pack_mat<<<(unsigned)((MT_N * 2 + 7) / 8), 256, 0, stream>>>(dD, Apk, (long)MT_N * 2, 2, N_A - 1);
  runMLP(mPosen, Apk, MT_N, node, nullptr);
  k_pack_mat<<<(unsigned)((MT_E * 1 + 7) / 8), 256, 0, stream>>>(dD, Apk, (long)MT_E, 1, 1);
  runMLP(mPosee, Apk, MT_E, edge, nullptr);
  k_pack_mat<<<(unsigned)((MT_G * 126 + 7) / 8), 256, 0, stream>>>(dD, Apk, (long)MT_G * 126, 126, E_G);
  runMLP(mPoseg, Apk, MT_G, glob, nullptr);

  // 12 message-passing steps (old node/edge/glob consumed before residual update)
  for (int st = 0; st < 12; ++st) {
    k_pack_edge<<<(unsigned)(((long)MT_E * 32 + 7) / 8), 256, 0, stream>>>(edge, node, glob, Apk);
    runMLP(mE[st], Apk, MT_E, e1, edge);               // edge += e1 fused
    k_sent<<<(NNODE * LAT + 255) / 256, 256, 0, stream>>>(e1, sent);
    k_recv<<<(NNODE * LAT + 255) / 256, 256, 0, stream>>>(e1, recv);
    k_pack_node<<<(unsigned)(((long)MT_N * 32 + 7) / 8), 256, 0, stream>>>(node, sent, recv, glob, Apk);
    runMLP(mN[st], Apk, MT_N, n1, node);               // node += n1 fused
    k_gsum<<<(B_G * LAT + 255) / 256, 256, 0, stream>>>(n1, gn);
    k_gsum<<<(B_G * LAT + 255) / 256, 256, 0, stream>>>(sent, ge);  // ge = per-graph edge sum
    k_pack_glob<<<(unsigned)(((long)MT_G * 24 + 7) / 8), 256, 0, stream>>>(glob, gn, ge, Apk);
    runMLP(mG[st], Apk, MT_G, g1v, glob);              // glob += g1 fused
  }

  // outputs: (node, edge, glob) concatenated
  float* out = (float*)d_out;
  k_copy<<<(NNODE * LAT + 255) / 256, 256, 0, stream>>>(out, node, (long)NNODE * LAT);
  k_copy<<<(unsigned)(((long)NE * LAT + 255) / 256), 256, 0, stream>>>(out + (long)NNODE * LAT, edge, (long)NE * LAT);
  k_copy<<<(B_G * LAT + 255) / 256, 256, 0, stream>>>(out + (long)NNODE * LAT + (long)NE * LAT, glob, (long)B_G * LAT);
}
